// VQVAE_43387759624265
// MI455X (gfx1250) — matrix operand order repaired and checked by reading, not compiled, not run on hardware
//
#include <hip/hip_runtime.h>
#include <math.h>
#include <stdint.h>

typedef __attribute__((ext_vector_type(16))) _Float16 v16h;
typedef __attribute__((ext_vector_type(8)))  _Float16 v8h;
typedef __attribute__((ext_vector_type(4)))  _Float16 v4h;
typedef __attribute__((ext_vector_type(8)))  float    v8f;

#define N_HEADS 4
#define LDP 40   // LDS row pitch in halves: 80B -> 16B-aligned b128 frags, no bank conflicts

typedef __attribute__((address_space(3))) const void* lds_cptr;
__device__ static inline unsigned ldsOffset(const void* p) {
  return (unsigned)(uintptr_t)(lds_cptr)p;
}
// gfx1250 async copy: memory -> LDS, 16B per lane, tracked by ASYNCcnt.
#define ASYNC_B128(ldsaddr, gptr)                                           \
  asm volatile("global_load_async_to_lds_b128 %0, %1, off"                  \
               :: "v"(ldsaddr), "v"(gptr) : "memory")
#define WAIT_ASYNC() asm volatile("s_wait_asynccnt 0x0" ::: "memory")

// ---------------------------------------------------------------------------
// Implicit-GEMM convolution on WMMA (f16 in, f32 accumulate).
// Block = 128 threads = 4 waves; tile = 64 pixels x 64 channels; each wave
// runs 4 x v_wmma_f32_16x16x32_f16 per K-step sharing one B fragment.
// XF16/YF16 select f32 boundaries (network input / final output / z).
// TMODE=1 = ConvTranspose2d(k=4,s=2,p=1) via lhs-dilation-2 + flipped weights.
// ---------------------------------------------------------------------------
struct ConvArgs {
  const void* X; const float* W; const float* Bias; const _Float16* Add;
  void* Y;
  int B, Cin, H, Wi, Cout, Hout, Wout, stride, pad;
  int wsh;      // log2(Wout)
  int mtShift;  // log2(HWo / 64)
};

template <int KS, int TMODE, int XF16, int YF16>
__global__ __launch_bounds__(128) void conv_wmma_kernel(ConvArgs a) {
  __shared__ _Float16 As[64][LDP];
  __shared__ _Float16 Bs[64][LDP];
  const _Float16* Xh = (const _Float16*)a.X;
  const float*    Xf = (const float*)a.X;
  _Float16* Yh = (_Float16*)a.Y;
  float*    Yf = (float*)a.Y;
  const int tid = threadIdx.x, wave = tid >> 5, lane = tid & 31;
  const int HWo = a.Hout * a.Wout;
  const int b = blockIdx.x >> a.mtShift;
  const int mbase = (blockIdx.x & ((1 << a.mtShift) - 1)) << 6;
  const int nbase = blockIdx.y << 6;
  constexpr int KSQ = KS * KS;
  const int KK = a.Cin * KSQ;
  const bool wVec = ((KK & 3) == 0);
  v8f acc0 = {}, acc1 = {}, acc2 = {}, acc3 = {};
  for (int kt = 0; kt < KK; kt += 32) {
    // ---- stage A tile (im2col gather, zero padding), 64x32 halves ----
    for (int i = tid; i < 2048; i += 128) {
      int m = i >> 5, kk = i & 31;
      int kg = kt + kk, p = mbase + m;  // p < HWo by exact tiling
      _Float16 v = (_Float16)0.f;
      if (kg < KK) {
        int cin = kg / KSQ, rem = kg % KSQ;   // KSQ is constexpr
        int ky = rem / KS, kx = rem % KS;
        int oy = p >> a.wsh, ox = p & (a.Wout - 1);
        long srcIdx = -1;
        if (!TMODE) {
          int iy = oy * a.stride - a.pad + ky;
          int ix = ox * a.stride - a.pad + kx;
          if (iy >= 0 && iy < a.H && ix >= 0 && ix < a.Wi)
            srcIdx = (((long)b * a.Cin + cin) * a.H + iy) * a.Wi + ix;
        } else {
          int vy = oy - 2 + ky, vx = ox - 2 + kx;
          if (vy >= 0 && vx >= 0 && !(vy & 1) && !(vx & 1)) {
            int iy = vy >> 1, ix = vx >> 1;
            if (iy < a.H && ix < a.Wi)
              srcIdx = (((long)b * a.Cin + cin) * a.H + iy) * a.Wi + ix;
          }
        }
        if (srcIdx >= 0) v = XF16 ? Xh[srcIdx] : (_Float16)Xf[srcIdx];
      }
      As[m][kk] = v;
    }
    // ---- stage B tile (f32 weights -> f16), 64x32 halves ----
    if (!TMODE && wVec) {
      for (int i = tid; i < 512; i += 128) {          // float4-vectorized
        int n = i >> 3, kq = (i & 7) << 2;
        int kg = kt + kq, ng = nbase + n;
        float4 w4 = make_float4(0.f, 0.f, 0.f, 0.f);
        if (ng < a.Cout && kg < KK)
          w4 = *(const float4*)(a.W + (long)ng * KK + kg);
        v4h h;
        h[0] = (_Float16)w4.x; h[1] = (_Float16)w4.y;
        h[2] = (_Float16)w4.z; h[3] = (_Float16)w4.w;
        *(v4h*)&Bs[n][kq] = h;
      }
    } else {
      for (int i = tid; i < 2048; i += 128) {
        int n = i >> 5, kk = i & 31;
        int kg = kt + kk, ng = nbase + n;
        float v = 0.f;
        if (kg < KK && ng < a.Cout) {
          int cin = kg / KSQ, rem = kg % KSQ;
          if (!TMODE) {
            v = a.W[((long)ng * a.Cin + cin) * KSQ + rem];
          } else {
            int ky = rem / KS, kx = rem % KS;
            v = a.W[((long)cin * a.Cout + ng) * 16 + (3 - ky) * 4 + (3 - kx)];
          }
        }
        Bs[n][kk] = (_Float16)v;
      }
    }
    if (!TMODE && kt + 32 < KK)                 // gfx1250 global_prefetch path
      __builtin_prefetch(a.W + (long)nbase * KK + kt + 32, 0, 1);
    __syncthreads();
    // ---- fragments per CDNA5 WMMA VGPR layout ----
    const int row = lane & 15, hi = lane >> 4;
    v8h blo = *(const v8h*)&Bs[wave * 16 + row][hi * 8];
    v8h bhi = *(const v8h*)&Bs[wave * 16 + row][hi * 8 + 16];
    v16h bf = __builtin_shufflevector(blo, bhi, 0, 1, 2, 3, 4, 5, 6, 7, 8, 9, 10, 11, 12, 13, 14, 15);
#define MK_AF(S)                                                            \
    v8h alo##S = *(const v8h*)&As[S * 16 + row][hi * 8];                    \
    v8h ahi##S = *(const v8h*)&As[S * 16 + row][hi * 8 + 16];               \
    v16h af##S = __builtin_shufflevector(alo##S, ahi##S, 0, 1, 2, 3, 4, 5,  \
                                         6, 7, 8, 9, 10, 11, 12, 13, 14, 15)
    MK_AF(0); MK_AF(1); MK_AF(2); MK_AF(3);
#undef MK_AF
    acc0 = __builtin_amdgcn_wmma_f32_16x16x32_f16(false, af0, false, bf, (short)0, acc0, false, false);
    acc1 = __builtin_amdgcn_wmma_f32_16x16x32_f16(false, af1, false, bf, (short)0, acc1, false, false);
    acc2 = __builtin_amdgcn_wmma_f32_16x16x32_f16(false, af2, false, bf, (short)0, acc2, false, false);
    acc3 = __builtin_amdgcn_wmma_f32_16x16x32_f16(false, af3, false, bf, (short)0, acc3, false, false);
    __syncthreads();
  }
  // ---- epilogue: pixels contiguous in NCHW; one v8h (or 2x float4) store ----
  const int n_g = nbase + wave * 16 + (lane & 15);
  if (n_g >= a.Cout) return;
  const float bias = a.Bias ? a.Bias[n_g] : 0.f;
  const int mrow0 = (lane >> 4) * 8;
  const long cbase = ((long)b * a.Cout + n_g) * HWo;
  const v8f accs[4] = {acc0, acc1, acc2, acc3};
  for (int s = 0; s < 4; ++s) {
    const long p0 = cbase + mbase + s * 16 + mrow0;
    float vals[8];
    for (int r = 0; r < 8; ++r) vals[r] = accs[s][r] + bias;
    if (a.Add) {
      v8h av = *(const v8h*)(a.Add + p0);
      for (int r = 0; r < 8; ++r) vals[r] += (float)av[r];
    }
    if (YF16) {
      v8h hv;
      for (int r = 0; r < 8; ++r) hv[r] = (_Float16)vals[r];
      *(v8h*)(Yh + p0) = hv;
    } else {
      float4 lo = make_float4(vals[0], vals[1], vals[2], vals[3]);
      float4 hq = make_float4(vals[4], vals[5], vals[6], vals[7]);
      *(float4*)(Yf + p0) = lo;
      *(float4*)(Yf + p0 + 4) = hq;
    }
  }
}

// ---------------------------------------------------------------------------
// Batched strided GEMM on WMMA. A is always f16 and staged via
// global_load_async_to_lds_b128 (ASYNCcnt). B: BF16=1 -> f16 (async when NxK),
// BF16=0 -> f32 weights (float4+cvt). All call sites have M,N % 64 == 0 and
// K % 32 == 0 -> guard-free.
// ---------------------------------------------------------------------------
struct GemmArgs {
  const void* A; const void* B; const float* Bias; const _Float16* Add;
  _Float16* C;
  int M, N, K, lda, ldb, transB, innerShift;
  long aOuter, aInner, bOuter, bInner, cOuter, cInner, cStrideM, cStrideN;
  float scale;
};

template <int BF16>
__global__ __launch_bounds__(128) void gemm_wmma_kernel(GemmArgs g) {
  __shared__ _Float16 As[64][LDP];
  __shared__ _Float16 Bs[64][LDP];
  const _Float16* Ah = (const _Float16*)g.A;
  const _Float16* Bh = (const _Float16*)g.B;
  const float*    Bf = (const float*)g.B;
  const int tid = threadIdx.x, wave = tid >> 5, lane = tid & 31;
  const int mbase = blockIdx.x << 6;
  const int nbase = blockIdx.y << 6;
  const int outer = blockIdx.z >> g.innerShift;
  const int inner = blockIdx.z & ((1 << g.innerShift) - 1);
  const long aOff = (long)outer * g.aOuter + (long)inner * g.aInner;
  const long bOff = (long)outer * g.bOuter + (long)inner * g.bInner;
  const long cOff = (long)outer * g.cOuter + (long)inner * g.cInner;
  const unsigned asBase = ldsOffset(&As[0][0]);
  const unsigned bsBase = ldsOffset(&Bs[0][0]);
  v8f acc0 = {}, acc1 = {}, acc2 = {}, acc3 = {};
  for (int kt = 0; kt < g.K; kt += 32) {
    // ---- A tile: async copy straight into LDS (2 x b128 per lane) ----
    for (int c = tid; c < 256; c += 128) {
      int m = c >> 2, q = c & 3;
      unsigned la = asBase + (unsigned)(m * (LDP * 2) + q * 16);
      const _Float16* gp = Ah + aOff + (long)(mbase + m) * g.lda + kt + q * 8;
      ASYNC_B128(la, gp);
    }
    if (BF16) {
      if (g.transB) {
        for (int c = tid; c < 256; c += 128) {      // B (NxK f16): async
          int n = c >> 2, q = c & 3;
          unsigned lb = bsBase + (unsigned)(n * (LDP * 2) + q * 16);
          const _Float16* gp = Bh + bOff + (long)(nbase + n) * g.ldb + kt + q * 8;
          ASYNC_B128(lb, gp);
        }
      } else {
        for (int i = tid; i < 2048; i += 128) {     // B (KxN f16): gather
          int n = i >> 5, kk = i & 31;
          Bs[n][kk] = Bh[bOff + (long)(kt + kk) * g.ldb + nbase + n];
        }
      }
    } else {
      for (int i = tid; i < 512; i += 128) {        // B (NxK f32 weights)
        int n = i >> 3, kq = (i & 7) << 2;
        float4 w4 = *(const float4*)(Bf + bOff + (long)(nbase + n) * g.ldb + kt + kq);
        v4h h;
        h[0] = (_Float16)w4.x; h[1] = (_Float16)w4.y;
        h[2] = (_Float16)w4.z; h[3] = (_Float16)w4.w;
        *(v4h*)&Bs[n][kq] = h;
      }
    }
    if (kt + 32 < g.K)
      __builtin_prefetch(Ah + aOff + (long)mbase * g.lda + kt + 32, 0, 1);
    WAIT_ASYNC();
    __syncthreads();
    const int row = lane & 15, hi = lane >> 4;
    v8h blo = *(const v8h*)&Bs[wave * 16 + row][hi * 8];
    v8h bhi = *(const v8h*)&Bs[wave * 16 + row][hi * 8 + 16];
    v16h bf = __builtin_shufflevector(blo, bhi, 0, 1, 2, 3, 4, 5, 6, 7, 8, 9, 10, 11, 12, 13, 14, 15);
#define MK_AF(S)                                                            \
    v8h alo##S = *(const v8h*)&As[S * 16 + row][hi * 8];                    \
    v8h ahi##S = *(const v8h*)&As[S * 16 + row][hi * 8 + 16];               \
    v16h af##S = __builtin_shufflevector(alo##S, ahi##S, 0, 1, 2, 3, 4, 5,  \
                                         6, 7, 8, 9, 10, 11, 12, 13, 14, 15)
    MK_AF(0); MK_AF(1); MK_AF(2); MK_AF(3);
#undef MK_AF
    acc0 = __builtin_amdgcn_wmma_f32_16x16x32_f16(false, af0, false, bf, (short)0, acc0, false, false);
    acc1 = __builtin_amdgcn_wmma_f32_16x16x32_f16(false, af1, false, bf, (short)0, acc1, false, false);
    acc2 = __builtin_amdgcn_wmma_f32_16x16x32_f16(false, af2, false, bf, (short)0, acc2, false, false);
    acc3 = __builtin_amdgcn_wmma_f32_16x16x32_f16(false, af3, false, bf, (short)0, acc3, false, false);
    __syncthreads();
  }
  const int n_g = nbase + wave * 16 + (lane & 15);
  const float bias = g.Bias ? g.Bias[n_g] : 0.f;
  const int mrow0 = (lane >> 4) * 8;
  const v8f accs[4] = {acc0, acc1, acc2, acc3};
  for (int s = 0; s < 4; ++s) {
    const int m0 = mbase + s * 16 + mrow0;
    for (int r = 0; r < 8; ++r) {
      long idx = cOff + (long)(m0 + r) * g.cStrideM + (long)n_g * g.cStrideN;
      float v = accs[s][r] * g.scale + bias;
      if (g.Add) v += (float)g.Add[idx];
      g.C[idx] = (_Float16)v;
    }
  }
}

// ---------------------------------------------------------------------------
// GroupNorm (32 groups) on f16 activations (f32 statistics), optional fused
// SiLU, optional transposed (B,HW,C) output for attention.
// ---------------------------------------------------------------------------
__global__ __launch_bounds__(256)
void groupnorm_kernel(const _Float16* X, const float* gam, const float* bta,
                      _Float16* Y, int C, int hwsh, int doSilu, int transOut) {
  __shared__ float r1[256], r2[256];
  const int HW = 1 << hwsh;
  const int cpg = C >> 5;
  const int b = blockIdx.x >> 5, grp = blockIdx.x & 31;
  const long base = ((long)b * C + (long)grp * cpg) << hwsh;
  const int n = cpg << hwsh;
  float s = 0.f, s2 = 0.f;
  for (int i = threadIdx.x; i < n; i += 256) {
    float v = (float)X[base + i];
    s += v; s2 += v * v;
  }
  r1[threadIdx.x] = s; r2[threadIdx.x] = s2;
  __syncthreads();
  for (int st = 128; st > 0; st >>= 1) {
    if (threadIdx.x < st) {
      r1[threadIdx.x] += r1[threadIdx.x + st];
      r2[threadIdx.x] += r2[threadIdx.x + st];
    }
    __syncthreads();
  }
  const float mean = r1[0] / (float)n;
  const float var = r2[0] / (float)n - mean * mean;
  const float rstd = rsqrtf(var + 1e-5f);
  for (int i = threadIdx.x; i < n; i += 256) {
    int cl = i >> hwsh, p = i & (HW - 1);
    int c = grp * cpg + cl;
    float v = ((float)X[base + i] - mean) * rstd * gam[c] + bta[c];
    if (doSilu) v = v / (1.f + __expf(-v));
    if (transOut) Y[((long)b * HW + p) * C + c] = (_Float16)v;
    else          Y[base + i] = (_Float16)v;
  }
}

__global__ __launch_bounds__(256) void softmax_rows(_Float16* S, int L) {
  __shared__ float red[256];
  const long base = (long)blockIdx.x * L;
  float m = -1e30f;
  for (int i = threadIdx.x; i < L; i += 256) m = fmaxf(m, (float)S[base + i]);
  red[threadIdx.x] = m; __syncthreads();
  for (int st = 128; st > 0; st >>= 1) {
    if (threadIdx.x < st)
      red[threadIdx.x] = fmaxf(red[threadIdx.x], red[threadIdx.x + st]);
    __syncthreads();
  }
  m = red[0]; __syncthreads();
  float sum = 0.f;
  for (int i = threadIdx.x; i < L; i += 256) {
    float e = __expf((float)S[base + i] - m);
    S[base + i] = (_Float16)e; sum += e;
  }
  red[threadIdx.x] = sum; __syncthreads();
  for (int st = 128; st > 0; st >>= 1) {
    if (threadIdx.x < st) red[threadIdx.x] += red[threadIdx.x + st];
    __syncthreads();
  }
  const float inv = 1.f / red[0];
  for (int i = threadIdx.x; i < L; i += 256)
    S[base + i] = (_Float16)((float)S[base + i] * inv);
}

// Vector quantize: nearest of 8192 3-dim codes; writes q (==q_st) NCHW f32
// (part of the kernel output) and accumulates squared error. HWz = 1024.
__global__ __launch_bounds__(128)
void quantize_kernel(const _Float16* H, const float* cb, float* Z,
                     float* sumAcc, int Npts) {
  const int i = blockIdx.x * 128 + threadIdx.x;
  if (i >= Npts) return;
  const int b = i >> 10, p = i & 1023;
  const float x0 = (float)H[((long)b * 3 + 0) * 1024 + p];
  const float x1 = (float)H[((long)b * 3 + 1) * 1024 + p];
  const float x2 = (float)H[((long)b * 3 + 2) * 1024 + p];
  float best = 1e30f; int bi = 0;
  for (int j = 0; j < 8192; ++j) {
    float d0 = cb[j * 3 + 0] - x0;
    float d1 = cb[j * 3 + 1] - x1;
    float d2 = cb[j * 3 + 2] - x2;
    float d = d0 * d0 + d1 * d1 + d2 * d2;
    if (d < best) { best = d; bi = j; }
  }
  const float q0 = cb[bi * 3 + 0], q1 = cb[bi * 3 + 1], q2 = cb[bi * 3 + 2];
  Z[((long)b * 3 + 0) * 1024 + p] = q0;
  Z[((long)b * 3 + 1) * 1024 + p] = q1;
  Z[((long)b * 3 + 2) * 1024 + p] = q2;
  const float e = (q0 - x0) * (q0 - x0) + (q1 - x1) * (q1 - x1) + (q2 - x2) * (q2 - x2);
  atomicAdd(sumAcc, e);
}

__global__ void zero_kernel(float* p, int n) {
  int i = blockIdx.x * blockDim.x + threadIdx.x;
  if (i < n) p[i] = 0.f;
}

__global__ void loss_finalize(const float* s, float* out2, float inv) {
  if (threadIdx.x == 0) { float v = s[0] * inv; out2[0] = v; out2[1] = v; }
}

// ---------------------------------------------------------------------------
// Host-side orchestration
// ---------------------------------------------------------------------------
struct PConv { const float* w; const float* b; };
struct PGN   { const float* g; const float* bta; };
struct PNCC  { PGN gn; PConv c; };
struct PMHA  { const float* in_w; const float* in_b; const float* out_w; const float* out_b; };
struct PRes  { PNCC f, s; PConv r; };
struct PDown { PRes r0, r1; const float* dw; const float* db; };
struct PMid  { PRes r0; PGN an0; PMHA a0; PRes r1; PGN an1; PMHA a1; PRes r2; };
struct PUp   { const float* uw; const float* ub; PRes r0, r1; };

static inline void convL(hipStream_t st, const void* X, const float* W,
                         const float* bias, const _Float16* add, void* Y,
                         int B, int Cin, int H, int Wi, int Cout,
                         int Hout, int Wout, int ks, int stride, int pad,
                         int tmode, int xf16, int yf16) {
  ConvArgs a{X, W, bias, add, Y, B, Cin, H, Wi, Cout, Hout, Wout, stride, pad, 0, 0};
  const int HWo = Hout * Wout;
  const int mtiles = HWo >> 6;                 // HWo is a multiple of 64
  a.wsh = __builtin_ctz(Wout);
  a.mtShift = __builtin_ctz(mtiles);
  dim3 grid((unsigned)(B * mtiles), (unsigned)((Cout + 63) / 64), 1);
  dim3 blk(128);
  if (tmode)            conv_wmma_kernel<4, 1, 1, 1><<<grid, blk, 0, st>>>(a);
  else if (ks == 4)     conv_wmma_kernel<4, 0, 1, 1><<<grid, blk, 0, st>>>(a);
  else if (ks == 1) {
    if (xf16)           conv_wmma_kernel<1, 0, 1, 1><<<grid, blk, 0, st>>>(a);
    else                conv_wmma_kernel<1, 0, 0, 1><<<grid, blk, 0, st>>>(a);
  } else {              // ks == 3
    if (!xf16)          conv_wmma_kernel<3, 0, 0, 1><<<grid, blk, 0, st>>>(a);
    else if (!yf16)     conv_wmma_kernel<3, 0, 1, 0><<<grid, blk, 0, st>>>(a);
    else                conv_wmma_kernel<3, 0, 1, 1><<<grid, blk, 0, st>>>(a);
  }
}

static inline void gemmL(hipStream_t st, const GemmArgs& g, int zTotal, int bf16) {
  dim3 grid((unsigned)(g.M >> 6), (unsigned)(g.N >> 6), (unsigned)zTotal);
  if (bf16) gemm_wmma_kernel<1><<<grid, dim3(128), 0, st>>>(g);
  else      gemm_wmma_kernel<0><<<grid, dim3(128), 0, st>>>(g);
}

static inline void gnL(hipStream_t st, const _Float16* X, const PGN& p, _Float16* Y,
                       int B, int C, int HW, int silu, int trans) {
  groupnorm_kernel<<<dim3((unsigned)(B * 32)), dim3(256), 0, st>>>(
      X, p.g, p.bta, Y, C, __builtin_ctz(HW), silu, trans);
}

static void resnetL(hipStream_t st, const _Float16* in, _Float16* out, _Float16* tmp,
                    int B, int Cin, int Cout, int H, int Wi, const PRes& r) {
  const int HW = H * Wi;
  gnL(st, in, r.f.gn, out, B, Cin, HW, 1, 0);
  convL(st, out, r.f.c.w, r.f.c.b, nullptr, tmp, B, Cin, H, Wi, Cout, H, Wi, 3, 1, 1, 0, 1, 1);
  gnL(st, tmp, r.s.gn, out, B, Cout, HW, 1, 0);
  convL(st, out, r.s.c.w, r.s.c.b, nullptr, tmp, B, Cout, H, Wi, Cout, H, Wi, 3, 1, 1, 0, 1, 1);
  convL(st, in, r.r.w, r.r.b, tmp, out, B, Cin, H, Wi, Cout, H, Wi, 1, 1, 0, 0, 1, 1);
}

static void attnL(hipStream_t st, const _Float16* x, _Float16* out, _Float16* sbuf,
                  const PGN& gn, const PMHA& a, int B, int C, int HW) {
  // scratch carved from the upper part of `out` (activations here are tiny)
  _Float16* G   = out + (size_t)2097152;  // (B*HW, C)
  _Float16* QKV = out + (size_t)4194304;  // (B*HW, 3C)
  _Float16* O   = out + (size_t)8388608;  // (B*HW, C)
  const int hd = C / N_HEADS;
  gnL(st, x, gn, G, B, C, HW, 0, 1);
  GemmArgs g{};
  // qkv = G @ in_w^T + in_b
  g.A = G; g.B = a.in_w; g.Bias = a.in_b; g.C = QKV;
  g.M = B * HW; g.N = 3 * C; g.K = C; g.lda = C; g.ldb = C;
  g.transB = 1; g.innerShift = 0;
  g.cStrideM = 3 * C; g.cStrideN = 1; g.scale = 1.f;
  gemmL(st, g, 1, 0);
  // S[b,h] = (Q K^T) / sqrt(hd)
  g = GemmArgs{};
  g.A = QKV; g.B = QKV + C; g.C = sbuf;
  g.M = HW; g.N = HW; g.K = hd; g.lda = 3 * C; g.ldb = 3 * C;
  g.transB = 1; g.innerShift = 2;
  g.aOuter = (long)HW * 3 * C; g.aInner = hd;
  g.bOuter = (long)HW * 3 * C; g.bInner = hd;
  g.cOuter = (long)N_HEADS * HW * HW; g.cInner = (long)HW * HW;
  g.cStrideM = HW; g.cStrideN = 1;
  g.scale = 1.0f / sqrtf((float)hd);
  gemmL(st, g, B * N_HEADS, 1);
  softmax_rows<<<dim3((unsigned)(B * N_HEADS * HW)), dim3(256), 0, st>>>(sbuf, HW);
  // O[b,h] = P V   (written interleaved into (B*HW, C))
  g = GemmArgs{};
  g.A = sbuf; g.B = QKV + 2 * C; g.C = O;
  g.M = HW; g.N = hd; g.K = HW; g.lda = HW; g.ldb = 3 * C;
  g.transB = 0; g.innerShift = 2;
  g.aOuter = (long)N_HEADS * HW * HW; g.aInner = (long)HW * HW;
  g.bOuter = (long)HW * 3 * C; g.bInner = hd;
  g.cOuter = (long)HW * C; g.cInner = hd;
  g.cStrideM = C; g.cStrideN = 1; g.scale = 1.f;
  gemmL(st, g, B * N_HEADS, 1);
  // out = x + O @ out_w^T + out_b, stored transposed back to NCHW
  g = GemmArgs{};
  g.A = O; g.B = a.out_w; g.Bias = a.out_b; g.Add = x; g.C = out;
  g.M = HW; g.N = C; g.K = C; g.lda = C; g.ldb = C;
  g.transB = 1; g.innerShift = 0;
  g.aOuter = (long)HW * C; g.cOuter = (long)C * HW;
  g.cStrideM = 1; g.cStrideN = HW; g.scale = 1.f;
  gemmL(st, g, B, 0);
}

extern "C" void kernel_launch(void* const* d_in, const int* in_sizes, int n_in,
                              void* d_out, int out_size, void* d_ws, size_t ws_size,
                              hipStream_t stream) {
  (void)in_sizes; (void)n_in; (void)out_size; (void)ws_size;
  const float* X = (const float*)d_in[0];
  int idx = 1;
  auto nx = [&]() { return (const float*)d_in[idx++]; };
  auto takeConv = [&]() { PConv c; c.w = nx(); c.b = nx(); return c; };
  auto takeGN   = [&]() { PGN g; g.g = nx(); g.bta = nx(); return g; };
  auto takeNCC  = [&]() { PNCC n; n.gn = takeGN(); n.c = takeConv(); return n; };
  auto takeMHA  = [&]() { PMHA m; m.in_w = nx(); m.in_b = nx();
                          m.out_w = nx(); m.out_b = nx(); return m; };
  auto takeRes  = [&]() { PRes r; r.f = takeNCC(); r.s = takeNCC(); r.r = takeConv(); return r; };
  auto takeDown = [&]() { PDown d; d.r0 = takeRes(); d.r1 = takeRes();
                          d.dw = nx(); d.db = nx(); return d; };
  auto takeMid  = [&]() { PMid m; m.r0 = takeRes(); m.an0 = takeGN(); m.a0 = takeMHA();
                          m.r1 = takeRes(); m.an1 = takeGN(); m.a1 = takeMHA();
                          m.r2 = takeRes(); return m; };
  auto takeUp   = [&]() { PUp u; u.uw = nx(); u.ub = nx();
                          u.r0 = takeRes(); u.r1 = takeRes(); return u; };

  PConv enc_in = takeConv();
  PDown down0 = takeDown();
  PDown down1 = takeDown();
  PDown down2 = takeDown();
  PMid enc_mid = takeMid();
  PGN  enc_norm = takeGN();
  PConv enc_out = takeConv();
  PConv pre_q = takeConv();
  const float* codebook = nx();
  PConv post_q = takeConv();
  PConv dec_in = takeConv();
  PMid dec_mid = takeMid();
  PUp up0 = takeUp();
  PUp up1 = takeUp();
  PUp up2 = takeUp();
  PGN dec_norm = takeGN();
  PConv dec_out = takeConv();

  // three 64MB f16 ping-pong activation buffers + scalar scratch
  const size_t SZ = 33554432;  // elements (f16)
  _Float16* B0 = (_Float16*)d_ws;
  _Float16* B1 = B0 + SZ;
  _Float16* B2 = B1 + SZ;
  float* sumAcc = (float*)(B2 + SZ);
  float* OUT = (float*)d_out;
  float* Z = OUT + 786432;         // 4*3*256*256
  float* LOSS = Z + 12288;         // 4*3*32*32

  _Float16* cur = B0; _Float16* o = B1; _Float16* t = B2;
  auto swp = [&]() { _Float16* x2 = cur; cur = o; o = x2; };
  auto doRes = [&](const PRes& r, int Cin, int Cout, int H) {
    resnetL(stream, cur, o, t, 4, Cin, Cout, H, H, r);
    swp();
  };
  auto doDownConv = [&](const float* w, const float* b, int C, int H) {
    convL(stream, cur, w, b, nullptr, o, 4, C, H, H, C, H / 2, H / 2, 4, 2, 1, 0, 1, 1);
    swp();
  };
  auto doAttn = [&](const PGN& gn, const PMHA& a, int C, int HW) {
    attnL(stream, cur, o, t, gn, a, 4, C, HW);
    swp();
  };
  auto doMid = [&](const PMid& m, int C, int H) {
    doRes(m.r0, C, C, H);
    doAttn(m.an0, m.a0, C, H * H);
    doRes(m.r1, C, C, H);
    doAttn(m.an1, m.a1, C, H * H);
    doRes(m.r2, C, C, H);
  };
  auto doUp = [&](const PUp& u, int Cin, int Cout, int H) {
    convL(stream, cur, u.uw, u.ub, nullptr, o, 4, Cin, H, H, Cin, 2 * H, 2 * H, 4, 1, 2, 1, 1, 1);
    swp();
    doRes(u.r0, Cin, Cout, 2 * H);
    doRes(u.r1, Cout, Cout, 2 * H);
  };

  // ---------------- encoder ----------------
  convL(stream, X, enc_in.w, enc_in.b, nullptr, cur, 4, 3, 256, 256, 64, 256, 256, 3, 1, 1, 0, 0, 1);
  doRes(down0.r0, 64, 128, 256);  doRes(down0.r1, 128, 128, 256);
  doDownConv(down0.dw, down0.db, 128, 256);
  doRes(down1.r0, 128, 256, 128); doRes(down1.r1, 256, 256, 128);
  doDownConv(down1.dw, down1.db, 256, 128);
  doRes(down2.r0, 256, 256, 64);  doRes(down2.r1, 256, 256, 64);
  doDownConv(down2.dw, down2.db, 256, 64);
  doMid(enc_mid, 256, 32);
  gnL(stream, cur, enc_norm, o, 4, 256, 1024, 1, 0); swp();
  convL(stream, cur, enc_out.w, enc_out.b, nullptr, o, 4, 256, 32, 32, 3, 32, 32, 3, 1, 1, 0, 1, 1); swp();
  convL(stream, cur, pre_q.w, pre_q.b, nullptr, o, 4, 3, 32, 32, 3, 32, 32, 1, 1, 0, 0, 1, 1); swp();

  // ---------------- quantize ----------------
  zero_kernel<<<dim3(1), dim3(32), 0, stream>>>(sumAcc, 1);
  quantize_kernel<<<dim3(32), dim3(128), 0, stream>>>(cur, codebook, Z, sumAcc, 4096);
  loss_finalize<<<dim3(1), dim3(32), 0, stream>>>(sumAcc, LOSS, 1.0f / 12288.0f);

  // ---------------- decoder ----------------
  convL(stream, Z, post_q.w, post_q.b, nullptr, o, 4, 3, 32, 32, 3, 32, 32, 1, 1, 0, 0, 0, 1); swp();
  convL(stream, cur, dec_in.w, dec_in.b, nullptr, o, 4, 3, 32, 32, 256, 32, 32, 3, 1, 1, 0, 1, 1); swp();
  doMid(dec_mid, 256, 32);
  doUp(up0, 256, 256, 32);
  doUp(up1, 256, 128, 64);
  doUp(up2, 128, 64, 128);
  gnL(stream, cur, dec_norm, o, 4, 64, 65536, 1, 0); swp();
  convL(stream, cur, dec_out.w, dec_out.b, nullptr, OUT, 4, 64, 256, 256, 3, 256, 256, 3, 1, 1, 0, 1, 0);
}
